// NeRFMLP_26027501814304
// MI455X (gfx1250) — matrix operand order repaired and checked by reading, not compiled, not run on hardware
//
#include <hip/hip_runtime.h>

// ---------------------------------------------------------------------------
// NeRF MLP forward on gfx1250 (MI455X), wave32 + v_wmma_f32_16x16x32_f16.
// Each wave processes 16 rows; all weights live in LDS as f16 (transposed).
// ---------------------------------------------------------------------------

typedef __attribute__((ext_vector_type(16))) _Float16 v16h;
typedef __attribute__((ext_vector_type(8)))  _Float16 v8h;
typedef __attribute__((ext_vector_type(8)))  float    v8f;

#define INPUT_DIM 32
#define HIDDEN    64
#define NB        4096
#define NS        128
#define NROWS     (NB * NS)        // 524288
#define NTILES    (NROWS / 16)     // 32768
#define WPB       4                // waves per block
#define NBLOCKS   2048
#define TPW       (NTILES / (NBLOCKS * WPB))   // 4 tiles per wave

struct alignas(16) SharedMem {
    // f16 transposed weights: WT[n][k] = W[k][n]
    _Float16 w0t[64 * 32];    // density L0: K=32
    _Float16 w1t[64 * 64];    // density L1: K=64
    _Float16 w2t[64 * 64];    // density L2: K=64
    _Float16 cw0t[64 * 96];   // color  L0: K=67 padded to 96 (zeros)
    _Float16 cw1t[64 * 64];   // color  L1: K=64
    _Float16 dow[64];         // density out  (64 -> 1)
    _Float16 cowt[3 * 64];    // color out^T  (64 -> 3), cowt[c][k]
    float db0[64], db1[64], db2[64], cb0[64], cb1[64];
    float dob;
    float cob[3];
    // per-wave activation ping-pong buffers
    _Float16 bufA[WPB][16 * 96];  // stride 96; cols 64..95 hold [vd|0] padding
    _Float16 bufB[WPB][16 * 64];  // stride 64
};

__device__ __forceinline__ void wavebar() { __builtin_amdgcn_wave_barrier(); }

// A fragment (16x32 f16, M x K):
//   lane<16 : row M=lane,    K = k0+{0..7} and k0+{16..23}
//   lane>=16: row M=lane-16, K = k0+{8..15} and k0+{24..31}
__device__ __forceinline__ v16h ldsA(const _Float16* base, int stride, int lane, int k0) {
    int m = lane & 15, h = lane >> 4;
    const _Float16* p = base + m * stride + k0 + h * 8;
    union { v16h v; v8h q[2]; } u;
    u.q[0] = *(const v8h*)(p);
    u.q[1] = *(const v8h*)(p + 16);
    return u.v;
}

// B fragment (32x16 f16, K x N) from row-major W^T[n][k]:
//   lane<16 : col N=n0+lane,    K = k0+{0..7} and k0+{16..23}
//   lane>=16: col N=n0+lane-16, K = k0+{8..15} and k0+{24..31}
__device__ __forceinline__ v16h ldsB(const _Float16* wt, int stride, int lane, int n0, int k0) {
    int n = n0 + (lane & 15), h = lane >> 4;
    const _Float16* p = wt + n * stride + k0 + h * 8;
    union { v16h v; v8h q[2]; } u;
    u.q[0] = *(const v8h*)(p);
    u.q[1] = *(const v8h*)(p + 16);
    return u.v;
}

// One 16-row x 64-out dense layer with ReLU: out = relu(in * W + b), f16 LDS.
// C/D layout: VGPR r, lane l -> M = r + (l>>4)*8, N = l&15 (+16*nt).
__device__ __forceinline__ void layer(const _Float16* in, int sin, int K,
                                      const _Float16* wt, int swt,
                                      const float* bias,
                                      _Float16* out, int sout, int lane) {
    v8f acc[4];
    for (int nt = 0; nt < 4; ++nt) {
        v8f c = {};
        for (int k0 = 0; k0 < K; k0 += 32) {
            v16h a = ldsA(in, sin, lane, k0);
            v16h b = ldsB(wt, swt, lane, nt * 16, k0);
            c = __builtin_amdgcn_wmma_f32_16x16x32_f16(false, a, false, b,
                                                       (short)0, c, false, false);
        }
        acc[nt] = c;
    }
    wavebar();
    int nl = lane & 15, mb = (lane >> 4) * 8;
    for (int nt = 0; nt < 4; ++nt) {
        int n = nt * 16 + nl;
        float bv = bias[n];
        for (int r = 0; r < 8; ++r) {
            float v = fmaxf(acc[nt][r] + bv, 0.0f);
            out[(mb + r) * sout + n] = (_Float16)v;
        }
    }
}

__global__ void __launch_bounds__(128)
nerf_mlp_kernel(const float* __restrict__ enc, const float* __restrict__ vdirs,
                const float* __restrict__ dW0, const float* __restrict__ dW1,
                const float* __restrict__ dW2,
                const float* __restrict__ db0, const float* __restrict__ db1,
                const float* __restrict__ db2,
                const float* __restrict__ doW, const float* __restrict__ dob,
                const float* __restrict__ cW0, const float* __restrict__ cW1,
                const float* __restrict__ cb0, const float* __restrict__ cb1,
                const float* __restrict__ coW, const float* __restrict__ cob,
                float* __restrict__ outDen, float* __restrict__ outRgb) {
    __shared__ SharedMem sm;
    const int tid = threadIdx.x;

    // ---- stage all weights (transposed, f16) + biases into LDS ----
    for (int i = tid; i < 64 * 32; i += 128) { int n = i >> 5, k = i & 31; sm.w0t[n * 32 + k] = (_Float16)dW0[k * 64 + n]; }
    for (int i = tid; i < 64 * 64; i += 128) { int n = i >> 6, k = i & 63; sm.w1t[n * 64 + k] = (_Float16)dW1[k * 64 + n]; }
    for (int i = tid; i < 64 * 64; i += 128) { int n = i >> 6, k = i & 63; sm.w2t[n * 64 + k] = (_Float16)dW2[k * 64 + n]; }
    for (int i = tid; i < 64 * 96; i += 128) {
        int n = i / 96, k = i % 96;
        sm.cw0t[n * 96 + k] = (k < 67) ? (_Float16)cW0[k * 64 + n] : (_Float16)0.0f;
    }
    for (int i = tid; i < 64 * 64; i += 128) { int n = i >> 6, k = i & 63; sm.cw1t[n * 64 + k] = (_Float16)cW1[k * 64 + n]; }
    for (int i = tid; i < 3 * 64; i += 128)  { int c = i >> 6, k = i & 63; sm.cowt[c * 64 + k] = (_Float16)coW[k * 3 + c]; }
    if (tid < 64) {
        sm.dow[tid] = (_Float16)doW[tid];
        sm.db0[tid] = db0[tid]; sm.db1[tid] = db1[tid]; sm.db2[tid] = db2[tid];
        sm.cb0[tid] = cb0[tid]; sm.cb1[tid] = cb1[tid];
    }
    if (tid == 0) { sm.dob = dob[0]; sm.cob[0] = cob[0]; sm.cob[1] = cob[1]; sm.cob[2] = cob[2]; }
    __syncthreads();

    const int wave = tid >> 5;
    const int lane = tid & 31;
    _Float16* bufA = sm.bufA[wave];
    _Float16* bufB = sm.bufB[wave];

    const int waveGlobal = blockIdx.x * WPB + wave;
    const int m = lane & 15, h = lane >> 4;

    for (int t = 0; t < TPW; ++t) {
        const int tile = waveGlobal * TPW + t;
        const int row0 = tile * 16;

        // ---- stage inputs: 16 rows x 32 f32 -> f16 bufB; view dirs into bufA[:,64..95]
        {
            const float4* s4 = (const float4*)(enc + (size_t)(row0 + m) * INPUT_DIM + h * 16);
            _Float16* dst = bufB + m * 64 + h * 16;
            for (int j = 0; j < 4; ++j) {
                float4 f = s4[j];
                dst[j * 4 + 0] = (_Float16)f.x; dst[j * 4 + 1] = (_Float16)f.y;
                dst[j * 4 + 2] = (_Float16)f.z; dst[j * 4 + 3] = (_Float16)f.w;
            }
            const int b = row0 / NS;   // 8 tiles per batch row -> uniform per tile
            _Float16* dv = bufA + m * 96 + 64 + h * 16;
            for (int j = 0; j < 16; ++j) {
                int col = h * 16 + j;  // relative to col 64
                float val = (col < 3) ? vdirs[b * 3 + col] : 0.0f;
                dv[j] = (_Float16)val;
            }
        }
        wavebar();

        // ---- density net ----
        layer(bufB, 64, 32, sm.w0t, 32, sm.db0, bufA, 96, lane);  wavebar();
        layer(bufA, 96, 64, sm.w1t, 64, sm.db1, bufB, 64, lane);  wavebar();
        layer(bufB, 64, 64, sm.w2t, 64, sm.db2, bufA, 96, lane);  wavebar();

        // ---- density head: dot(h, doW) + dob (2 lanes per row, 32 MACs each)
        {
            float s = 0.0f;
            const _Float16* hp = bufA + m * 96 + h * 32;
            const _Float16* wp = sm.dow + h * 32;
            for (int k = 0; k < 32; ++k) s += (float)hp[k] * (float)wp[k];
            s += __shfl_xor(s, 16, 32);
            if (h == 0) outDen[row0 + m] = s + sm.dob;
        }
        wavebar();

        // ---- color net: input [h | vd | 0] is bufA (16x96) ----
        layer(bufA, 96, 96, sm.cw0t, 96, sm.cb0, bufB, 64, lane); wavebar();
        layer(bufB, 64, 64, sm.cw1t, 64, sm.cb1, bufA, 96, lane); wavebar();

        // ---- color head: sigmoid(dot(h2, coW[:,c]) + cob[c]), c = 0..2 ----
        {
            float s0 = 0.f, s1 = 0.f, s2 = 0.f;
            const _Float16* hp = bufA + m * 96 + h * 32;
            const _Float16* w0 = sm.cowt + 0 * 64 + h * 32;
            const _Float16* w1 = sm.cowt + 1 * 64 + h * 32;
            const _Float16* w2 = sm.cowt + 2 * 64 + h * 32;
            for (int k = 0; k < 32; ++k) {
                float x = (float)hp[k];
                s0 += x * (float)w0[k]; s1 += x * (float)w1[k]; s2 += x * (float)w2[k];
            }
            s0 += __shfl_xor(s0, 16, 32);
            s1 += __shfl_xor(s1, 16, 32);
            s2 += __shfl_xor(s2, 16, 32);
            if (h == 0) {
                float* o = outRgb + (size_t)(row0 + m) * 3;
                o[0] = 1.0f / (1.0f + __expf(-(s0 + sm.cob[0])));
                o[1] = 1.0f / (1.0f + __expf(-(s1 + sm.cob[1])));
                o[2] = 1.0f / (1.0f + __expf(-(s2 + sm.cob[2])));
            }
        }
        wavebar();
    }
}

extern "C" void kernel_launch(void* const* d_in, const int* in_sizes, int n_in,
                              void* d_out, int out_size, void* d_ws, size_t ws_size,
                              hipStream_t stream) {
    (void)in_sizes; (void)n_in; (void)d_ws; (void)ws_size; (void)out_size;
    // setup_inputs() dict order, tuples flattened:
    const float* enc = (const float*)d_in[0];   // [B,S,32]
    const float* vd  = (const float*)d_in[1];   // [B,3]
    const float* dW0 = (const float*)d_in[2];   // [32,64]
    const float* dW1 = (const float*)d_in[3];   // [64,64]
    const float* dW2 = (const float*)d_in[4];   // [64,64]
    const float* db0 = (const float*)d_in[5];
    const float* db1 = (const float*)d_in[6];
    const float* db2 = (const float*)d_in[7];
    const float* doW = (const float*)d_in[8];   // [64,1]
    const float* dob = (const float*)d_in[9];   // [1]
    const float* cW0 = (const float*)d_in[10];  // [67,64]
    const float* cW1 = (const float*)d_in[11];  // [64,64]
    const float* cb0 = (const float*)d_in[12];
    const float* cb1 = (const float*)d_in[13];
    const float* coW = (const float*)d_in[14];  // [64,3]
    const float* cob = (const float*)d_in[15];  // [3]

    float* outDen = (float*)d_out;              // [B,S,1] flat
    float* outRgb = outDen + (size_t)NROWS;     // [B,S,3] flat

    nerf_mlp_kernel<<<NBLOCKS, 128, 0, stream>>>(enc, vd, dW0, dW1, dW2,
                                                 db0, db1, db2, doW, dob,
                                                 cW0, cW1, cb0, cb1, coW, cob,
                                                 outDen, outRgb);
}
